// MyGPT_69372311765566
// MI455X (gfx1250) — matrix operand order repaired and checked by reading, not compiled, not run on hardware
//
#include <hip/hip_runtime.h>
#include <hip/hip_bf16.h>

typedef __attribute__((ext_vector_type(16))) __bf16 v16bf;
typedef __attribute__((ext_vector_type(8)))  float  v8f;
typedef int v4i __attribute__((vector_size(16)));

#define LAYERS 8
#define SEQ    2048
#define DMODEL 1024
#define NHEAD  16
#define HDIM   64
#define FFN    (4 * DMODEL)
#define VOCAB  32000
#define BATCH  2
#define ROWS   (BATCH * SEQ)   // 4096

// ---- CDNA5 async global->LDS path (ASYNCcnt), with safe fallback ------------
#if defined(__has_builtin)
#if __has_builtin(__builtin_amdgcn_global_load_async_to_lds_b128)
#define ASYNC_LDS 1
#endif
#endif

#ifdef ASYNC_LDS
typedef __attribute__((address_space(1))) v4i as1_v4i;
typedef __attribute__((address_space(3))) v4i as3_v4i;
#endif

__device__ __forceinline__ void stage16(const __bf16* g, __bf16* l) {
#ifdef ASYNC_LDS
    __builtin_amdgcn_global_load_async_to_lds_b128((as1_v4i*)g, (as3_v4i*)l, 0, 0);
#else
    *reinterpret_cast<uint4*>(l) = *reinterpret_cast<const uint4*>(g);
#endif
}

__device__ __forceinline__ void stage_wait() {
#ifdef ASYNC_LDS
#if __has_builtin(__builtin_amdgcn_s_wait_asynccnt)
    __builtin_amdgcn_s_wait_asynccnt(0);
#else
    asm volatile("s_wait_asynccnt 0x0" ::: "memory");
#endif
#endif
}

__device__ __forceinline__ float round4f(float v) {
    // jnp.round(x, 4): round-half-to-even at 1e-4 granularity
    return __builtin_rintf(v * 1.0e4f) * 1.0e-4f;
}

__device__ __forceinline__ v8f wmma_bf16(v16bf a, v16bf b, v8f c) {
    return __builtin_amdgcn_wmma_f32_16x16x32_bf16(false, a, false, b,
                                                   (short)0, c, false, false);
}

// ---- WMMA fragment loaders (ISA 7.12.2 layouts), from LDS -------------------
// A: 16x32 bf16, row-major tile. lane&15 = row M; lane>=16 selects K-half.
__device__ __forceinline__ v16bf load_a_frag(const __bf16* s, int ld, int row0, int lane) {
    const int m  = row0 + (lane & 15);
    const int kb = (lane >> 4) * 8;
    v16bf a;
#pragma unroll
    for (int e = 0; e < 16; ++e) {
        const int k = (e < 8) ? (kb + e) : (16 + kb + (e - 8));
        a[e] = s[m * ld + k];
    }
    return a;
}
// B: 32x16 bf16 from an n-major (transposed) tile: s[n][k].
__device__ __forceinline__ v16bf load_bT_frag(const __bf16* s, int ld, int col0, int lane) {
    const int n  = col0 + (lane & 15);
    const int kb = (lane >> 4) * 8;
    v16bf b;
#pragma unroll
    for (int e = 0; e < 16; ++e) b[e] = s[n * ld + kb + ((e < 8) ? e : (8 + e))];
    return b;
}

// ---- Embedding: x = wte[ctx] + wpe ------------------------------------------
__global__ void embed_kernel(const int* __restrict__ ctx, const float* __restrict__ wte,
                             const float* __restrict__ wpe, float* __restrict__ x) {
    size_t idx   = (size_t)blockIdx.x * blockDim.x + threadIdx.x;
    size_t total = (size_t)ROWS * DMODEL;
    size_t step  = (size_t)gridDim.x * blockDim.x;
    for (; idx < total; idx += step) {
        size_t rs = idx / DMODEL;
        int d = (int)(idx % DMODEL);
        int s = (int)(rs % SEQ);
        int tok = ctx[rs];
        x[idx] = wte[(size_t)tok * DMODEL + d] + wpe[(size_t)s * DMODEL + d];
    }
}

// ---- Weight conversions to bf16, stored TRANSPOSED (N x K) ------------------
// in [L,H,D,HD] -> out [L, N=D (c=h*64+k), K=D (d)]
__global__ void conv_head_T_kernel(const float* __restrict__ in, __bf16* __restrict__ out) {
    size_t idx   = (size_t)blockIdx.x * blockDim.x + threadIdx.x;
    size_t total = (size_t)LAYERS * DMODEL * DMODEL;
    size_t step  = (size_t)gridDim.x * blockDim.x;
    for (; idx < total; idx += step) {
        int l   = (int)(idx / ((size_t)DMODEL * DMODEL));
        int rem = (int)(idx % ((size_t)DMODEL * DMODEL));
        int c = rem / DMODEL, d = rem % DMODEL;   // c = output column, d = K index
        int h = c >> 6, kk = c & 63;
        out[idx] = (__bf16)in[(((size_t)l * NHEAD + h) * DMODEL + d) * HDIM + kk];
    }
}

// in [Lc, Kc, Nc] -> out [Lc, Nc, Kc]
__global__ void conv_T_kernel(const float* __restrict__ in, __bf16* __restrict__ out,
                              int Lc, int Kc, int Nc) {
    size_t total = (size_t)Lc * Kc * Nc;
    size_t idx   = (size_t)blockIdx.x * blockDim.x + threadIdx.x;
    size_t step  = (size_t)gridDim.x * blockDim.x;
    size_t plane = (size_t)Kc * Nc;
    for (; idx < total; idx += step) {
        int l   = (int)(idx / plane);
        size_t rem = idx - (size_t)l * plane;
        int n  = (int)(rem / Kc);
        int kd = (int)(rem % Kc);
        out[idx] = (__bf16)in[((size_t)l * Kc + kd) * Nc + n];
    }
}

// ---- LayerNorm -> bf16 ------------------------------------------------------
__global__ __launch_bounds__(256)
void ln_bf16_kernel(const float* __restrict__ x, const float* __restrict__ g,
                    const float* __restrict__ bta, __bf16* __restrict__ out) {
    __shared__ float red[256];
    const int row = blockIdx.x, tid = threadIdx.x;
    const float* xr = x + (size_t)row * DMODEL;
    float s = 0.f, ss = 0.f;
    for (int i = tid; i < DMODEL; i += 256) { float v = xr[i]; s += v; ss += v * v; }
    red[tid] = s; __syncthreads();
    for (int d = 128; d > 0; d >>= 1) { if (tid < d) red[tid] += red[tid + d]; __syncthreads(); }
    float mean = red[0] * (1.0f / DMODEL);
    __syncthreads();
    red[tid] = ss; __syncthreads();
    for (int d = 128; d > 0; d >>= 1) { if (tid < d) red[tid] += red[tid + d]; __syncthreads(); }
    float var = red[0] * (1.0f / DMODEL) - mean * mean;
    float inv = rsqrtf(var + 1e-5f);
    for (int i = tid; i < DMODEL; i += 256)
        out[(size_t)row * DMODEL + i] = (__bf16)((xr[i] - mean) * inv * g[i] + bta[i]);
}

// ---- Tiled bf16 WMMA GEMM: C[M,N] = A[M,K] * BT[N,K]^T ----------------------
// EPI 0: outF = C + bias            (logits)
// EPI 1: outB = bf16(C)             (q/k)
// EPI 2: outB = bf16(relu(C+bias))  (FFN1)
// EPI 3: outF = round4(C+bias+res)  (FFN2, in-place residual)
// EPI 4: outB[n*M+m] = bf16(C)      (v, transposed for attention)
template <int EPI>
__global__ __launch_bounds__(128)
void gemm_bf16_wmma(const __bf16* __restrict__ A, const __bf16* __restrict__ BT,
                    const float* __restrict__ bias, const float* __restrict__ resid,
                    float* __restrict__ outF, __bf16* __restrict__ outB,
                    int M, int N, int K, float scale) {
    __shared__ __bf16 As[64][40];    // 64(m) x 32(k) + pad
    __shared__ __bf16 BsT[64][40];   // 64(n) x 32(k) + pad
    const int tid  = threadIdx.x;
    const int lane = tid & 31;
    const int w    = tid >> 5;
    const int m0   = blockIdx.y * 64;
    const int n0   = blockIdx.x * 64;
    const int mr   = (w >> 1) * 32;
    const int nc   = (w & 1) * 32;

    v8f acc[2][2];
#pragma unroll
    for (int i = 0; i < 2; ++i)
#pragma unroll
        for (int j = 0; j < 2; ++j)
#pragma unroll
            for (int e = 0; e < 8; ++e) acc[i][j][e] = 0.0f;

    const int rr = tid >> 1, cc = (tid & 1) * 16;   // 2 threads/row, 16 bf16 each

    for (int k = 0; k < K; k += 32) {
        stage16(&A[(size_t)(m0 + rr) * K + k + cc], &As[rr][cc]);
        stage16(&BT[(size_t)(n0 + rr) * K + k + cc], &BsT[rr][cc]);
        if (k + 32 < K) {
            __builtin_prefetch(&A[(size_t)(m0 + rr) * K + (k + 32) + cc], 0, 1);
            __builtin_prefetch(&BT[(size_t)(n0 + rr) * K + (k + 32) + cc], 0, 1);
        }
        stage_wait();
        __syncthreads();
        v16bf a0 = load_a_frag(&As[0][0], 40, mr, lane);
        v16bf a1 = load_a_frag(&As[0][0], 40, mr + 16, lane);
        v16bf b0 = load_bT_frag(&BsT[0][0], 40, nc, lane);
        v16bf b1 = load_bT_frag(&BsT[0][0], 40, nc + 16, lane);
        acc[0][0] = wmma_bf16(a0, b0, acc[0][0]);
        acc[0][1] = wmma_bf16(a0, b1, acc[0][1]);
        acc[1][0] = wmma_bf16(a1, b0, acc[1][0]);
        acc[1][1] = wmma_bf16(a1, b1, acc[1][1]);
        __syncthreads();
    }

    const int hi = lane >> 4, nlo = lane & 15;
#pragma unroll
    for (int i = 0; i < 2; ++i)
#pragma unroll
        for (int j = 0; j < 2; ++j)
#pragma unroll
            for (int r = 0; r < 8; ++r) {
                int m = m0 + mr + 16 * i + r + 8 * hi;
                int n = n0 + nc + 16 * j + nlo;
                float vv = acc[i][j][r] * scale;
                size_t off = (size_t)m * N + n;
                if (EPI == 0) {
                    outF[off] = vv + bias[n];
                } else if (EPI == 1) {
                    outB[off] = (__bf16)vv;
                } else if (EPI == 2) {
                    float t = vv + bias[n];
                    outB[off] = (__bf16)(t > 0.0f ? t : 0.0f);
                } else if (EPI == 3) {
                    outF[off] = round4f(vv + bias[n] + resid[off]);
                } else {
                    outB[(size_t)n * M + m] = (__bf16)vv;   // transposed store
                }
            }
}

// ---- Flash attention: per-wave 16 queries, 32-key chunks, fused residual ----
// q,k row-major [ROWS, D]; vT transposed [D, ROWS].
__global__ __launch_bounds__(128)
void attn_kernel(const __bf16* __restrict__ q, const __bf16* __restrict__ kp,
                 const __bf16* __restrict__ vT, float* __restrict__ x) {
    __shared__ __bf16 Kt[32][72];    // 32(key) x 64(hd) + pad
    __shared__ __bf16 VtT[64][40];   // 64(hd) x 32(key) + pad
    __shared__ __bf16 Pt[4][16][40];
    const int tid = threadIdx.x, lane = tid & 31, w = tid >> 5;
    const int b  = blockIdx.y / NHEAD, hh = blockIdx.y % NHEAD;
    const int q0 = blockIdx.x * 64;
    const int qw = q0 + w * 16;
    const int hi = lane >> 4, nlo = lane & 15;
    const size_t rowbase = (size_t)b * SEQ;

    // Q fragments: 16x64 in A layout, split into 2 K=32 chunks
    v16bf qf[2];
    {
        const int m = qw + nlo;
        const int kb = hi * 8;
#pragma unroll
        for (int t = 0; t < 2; ++t)
#pragma unroll
            for (int e = 0; e < 16; ++e) {
                int kk = t * 32 + ((e < 8) ? (kb + e) : (16 + kb + (e - 8)));
                qf[t][e] = q[(rowbase + m) * DMODEL + hh * HDIM + kk];
            }
    }

    float rowmax[8], rowsum[8];
    v8f o[4];
#pragma unroll
    for (int r = 0; r < 8; ++r) { rowmax[r] = -1.0e30f; rowsum[r] = 0.0f; }
#pragma unroll
    for (int nt = 0; nt < 4; ++nt)
#pragma unroll
        for (int e = 0; e < 8; ++e) o[nt][e] = 0.0f;

    const int kr = tid >> 2, ks = (tid & 3) * 16;   // K tile: 4 threads/key-row
    const int vr = tid >> 1, vs = (tid & 1) * 16;   // V tile: 2 threads/hd-row

    for (int kc = 0; kc <= q0 + 63; kc += 32) {
        stage16(&kp[(rowbase + kc + kr) * DMODEL + hh * HDIM + ks], &Kt[kr][ks]);
        stage16(&vT[(size_t)(hh * HDIM + vr) * ROWS + rowbase + kc + vs], &VtT[vr][vs]);
        stage_wait();
        __syncthreads();
        if (kc <= qw + 15) {
            // S = Q K^T : two 16x16 score tiles over K=64 (2 WMMA steps each)
            v8f c0, c1;
#pragma unroll
            for (int e = 0; e < 8; ++e) { c0[e] = 0.0f; c1[e] = 0.0f; }
#pragma unroll
            for (int t = 0; t < 2; ++t) {
                v16bf bk0, bk1;
#pragma unroll
                for (int e = 0; e < 16; ++e) {
                    int hd = t * 32 + hi * 8 + ((e < 8) ? e : (8 + e));
                    bk0[e] = Kt[nlo][hd];
                    bk1[e] = Kt[16 + nlo][hd];
                }
                c0 = wmma_bf16(qf[t], bk0, c0);
                c1 = wmma_bf16(qf[t], bk1, c1);
            }
            // scale, causal mask, online softmax (row M = r + 8*hi per 16-lane group)
#pragma unroll
            for (int r = 0; r < 8; ++r) {
                const int i = qw + r + 8 * hi;
                float s0 = c0[r] * 0.125f;
                float s1 = c1[r] * 0.125f;
                if (kc + nlo > i)      s0 = -1.0e30f;
                if (kc + 16 + nlo > i) s1 = -1.0e30f;
                float mx = fmaxf(s0, s1);
#pragma unroll
                for (int d = 1; d < 16; d <<= 1) mx = fmaxf(mx, __shfl_xor(mx, d, 32));
                float nm = fmaxf(rowmax[r], mx);
                float sc = __expf(rowmax[r] - nm);
                float p0 = __expf(s0 - nm);
                float p1 = __expf(s1 - nm);
                float lsum = p0 + p1;
#pragma unroll
                for (int d = 1; d < 16; d <<= 1) lsum += __shfl_xor(lsum, d, 32);
                rowsum[r] = rowsum[r] * sc + lsum;
                rowmax[r] = nm;
#pragma unroll
                for (int nt = 0; nt < 4; ++nt) o[nt][r] *= sc;
                const int mrow = r + 8 * hi;
                Pt[w][mrow][nlo]      = (__bf16)p0;
                Pt[w][mrow][16 + nlo] = (__bf16)p1;
            }
            // O += P V : A = P (16x32), B = V chunk from n-major VtT
            v16bf pa;
            {
                const int kb = hi * 8;
#pragma unroll
                for (int e = 0; e < 16; ++e) {
                    int kk = (e < 8) ? (kb + e) : (16 + kb + (e - 8));
                    pa[e] = Pt[w][nlo][kk];
                }
            }
#pragma unroll
            for (int nt = 0; nt < 4; ++nt) {
                v16bf bv;
#pragma unroll
                for (int e = 0; e < 16; ++e) bv[e] = VtT[nt * 16 + nlo][hi * 8 + ((e < 8) ? e : (8 + e))];
                o[nt] = wmma_bf16(pa, bv, o[nt]);
            }
        }
        __syncthreads();
    }
    // x += round4(O / rowsum)
#pragma unroll
    for (int nt = 0; nt < 4; ++nt)
#pragma unroll
        for (int r = 0; r < 8; ++r) {
            int i = qw + r + 8 * hi;
            int col = hh * HDIM + nt * 16 + nlo;
            size_t off = (rowbase + i) * DMODEL + col;
            x[off] += round4f(o[nt][r] / rowsum[r]);
        }
}

// ---- Row softmax over VOCAB with round4, in place ---------------------------
__global__ __launch_bounds__(256)
void softmax_round_kernel(float* __restrict__ out) {
    __shared__ float red[256];
    const int row = blockIdx.x, tid = threadIdx.x;
    float* r = out + (size_t)row * VOCAB;
    float mx = -1.0e30f;
    for (int i = tid; i < VOCAB; i += 256) mx = fmaxf(mx, r[i]);
    red[tid] = mx; __syncthreads();
    for (int d = 128; d > 0; d >>= 1) { if (tid < d) red[tid] = fmaxf(red[tid], red[tid + d]); __syncthreads(); }
    mx = red[0]; __syncthreads();
    float s = 0.0f;
    for (int i = tid; i < VOCAB; i += 256) { float e = __expf(r[i] - mx); r[i] = e; s += e; }
    red[tid] = s; __syncthreads();
    for (int d = 128; d > 0; d >>= 1) { if (tid < d) red[tid] += red[tid + d]; __syncthreads(); }
    float inv = 1.0f / red[0];
    __syncthreads();
    for (int i = tid; i < VOCAB; i += 256) r[i] = round4f(r[i] * inv);
}

// ---- Host orchestration ------------------------------------------------------
extern "C" void kernel_launch(void* const* d_in, const int* in_sizes, int n_in,
                              void* d_out, int out_size, void* d_ws, size_t ws_size,
                              hipStream_t stream) {
    const int*   ctx   = (const int*)d_in[0];
    const float* wte   = (const float*)d_in[1];
    const float* wpe   = (const float*)d_in[2];
    const float* ln1_s = (const float*)d_in[3];
    const float* ln1_b = (const float*)d_in[4];
    const float* Wq    = (const float*)d_in[5];
    const float* Wk    = (const float*)d_in[6];
    const float* Wv    = (const float*)d_in[7];
    const float* ln2_s = (const float*)d_in[8];
    const float* ln2_b = (const float*)d_in[9];
    const float* W1    = (const float*)d_in[10];
    const float* b1    = (const float*)d_in[11];
    const float* W2    = (const float*)d_in[12];
    const float* b2    = (const float*)d_in[13];
    const float* lnf_s = (const float*)d_in[14];
    const float* lnf_b = (const float*)d_in[15];
    const float* Wout  = (const float*)d_in[16];
    const float* bout  = (const float*)d_in[17];
    float* out = (float*)d_out;

    char* p = (char*)d_ws;
    auto carve = [&](size_t bytes) -> char* {
        char* r = p;
        p += (bytes + 255) & ~(size_t)255;
        return r;
    };
    __bf16* WqB   = (__bf16*)carve((size_t)LAYERS * DMODEL * DMODEL * 2);
    __bf16* WkB   = (__bf16*)carve((size_t)LAYERS * DMODEL * DMODEL * 2);
    __bf16* WvB   = (__bf16*)carve((size_t)LAYERS * DMODEL * DMODEL * 2);
    __bf16* W1B   = (__bf16*)carve((size_t)LAYERS * DMODEL * FFN * 2);
    __bf16* W2B   = (__bf16*)carve((size_t)LAYERS * FFN * DMODEL * 2);
    __bf16* WoutB = (__bf16*)carve((size_t)DMODEL * VOCAB * 2);
    float*  xbuf  = (float*)carve((size_t)ROWS * DMODEL * 4);
    __bf16* hbuf  = (__bf16*)carve((size_t)ROWS * DMODEL * 2);
    __bf16* qB    = (__bf16*)carve((size_t)ROWS * DMODEL * 2);
    __bf16* kB    = (__bf16*)carve((size_t)ROWS * DMODEL * 2);
    __bf16* vBT   = (__bf16*)carve((size_t)ROWS * DMODEL * 2);
    __bf16* ffh   = (__bf16*)carve((size_t)ROWS * FFN * 2);

    embed_kernel<<<2048, 256, 0, stream>>>(ctx, wte, wpe, xbuf);
    conv_head_T_kernel<<<2048, 256, 0, stream>>>(Wq, WqB);
    conv_head_T_kernel<<<2048, 256, 0, stream>>>(Wk, WkB);
    conv_head_T_kernel<<<2048, 256, 0, stream>>>(Wv, WvB);
    conv_T_kernel<<<4096, 256, 0, stream>>>(W1, W1B, LAYERS, DMODEL, FFN);
    conv_T_kernel<<<4096, 256, 0, stream>>>(W2, W2B, LAYERS, FFN, DMODEL);
    conv_T_kernel<<<4096, 256, 0, stream>>>(Wout, WoutB, 1, DMODEL, VOCAB);

    dim3 gQKV(DMODEL / 64, ROWS / 64);
    dim3 gFF1(FFN / 64, ROWS / 64);
    dim3 gFF2(DMODEL / 64, ROWS / 64);
    dim3 gOut(VOCAB / 64, ROWS / 64);
    dim3 gAtt(SEQ / 64, BATCH * NHEAD);

    for (int l = 0; l < LAYERS; ++l) {
        ln_bf16_kernel<<<ROWS, 256, 0, stream>>>(xbuf, ln1_s + (size_t)l * DMODEL,
                                                 ln1_b + (size_t)l * DMODEL, hbuf);
        gemm_bf16_wmma<1><<<gQKV, 128, 0, stream>>>(hbuf, WqB + (size_t)l * DMODEL * DMODEL,
            nullptr, nullptr, nullptr, qB, ROWS, DMODEL, DMODEL, 1.0f);
        gemm_bf16_wmma<1><<<gQKV, 128, 0, stream>>>(hbuf, WkB + (size_t)l * DMODEL * DMODEL,
            nullptr, nullptr, nullptr, kB, ROWS, DMODEL, DMODEL, 1.0f);
        gemm_bf16_wmma<4><<<gQKV, 128, 0, stream>>>(hbuf, WvB + (size_t)l * DMODEL * DMODEL,
            nullptr, nullptr, nullptr, vBT, ROWS, DMODEL, DMODEL, 1.0f);
        attn_kernel<<<gAtt, 128, 0, stream>>>(qB, kB, vBT, xbuf);
        ln_bf16_kernel<<<ROWS, 256, 0, stream>>>(xbuf, ln2_s + (size_t)l * DMODEL,
                                                 ln2_b + (size_t)l * DMODEL, hbuf);
        gemm_bf16_wmma<2><<<gFF1, 128, 0, stream>>>(hbuf, W1B + (size_t)l * DMODEL * FFN,
            b1 + (size_t)l * FFN, nullptr, nullptr, ffh, ROWS, FFN, DMODEL, 1.0f);
        gemm_bf16_wmma<3><<<gFF2, 128, 0, stream>>>(ffh, W2B + (size_t)l * FFN * DMODEL,
            b2 + (size_t)l * DMODEL, xbuf, xbuf, nullptr, ROWS, DMODEL, FFN, 1.0f);
    }
    ln_bf16_kernel<<<ROWS, 256, 0, stream>>>(xbuf, lnf_s, lnf_b, hbuf);
    gemm_bf16_wmma<0><<<gOut, 128, 0, stream>>>(hbuf, WoutB, bout, nullptr, out, nullptr,
                                                ROWS, VOCAB, DMODEL, 1.0f);
    softmax_round_kernel<<<ROWS, 256, 0, stream>>>(out);
}
